// Policy_58471684768145
// MI455X (gfx1250) — matrix operand order repaired and checked by reading, not compiled, not run on hardware
//
#include <hip/hip_runtime.h>

// ---------------- problem constants ----------------
#define NN   10000            // nodes
#define FIN  512              // input features
#define EDIM 256              // embedding dim
#define H0D  512
#define H1D  256
#define NE   160000           // edges
#define TT   (NE + NN)        // 170000 total MLP rows
#define NP   10016            // nodes padded to 16
#define TP   170016           // T padded to 32 (5313*32)

typedef __attribute__((ext_vector_type(16))) __bf16 v16bf;
typedef __attribute__((ext_vector_type(8)))  float  v8f;

// ---------------- bf16 helpers ----------------
__device__ __forceinline__ unsigned short f2bf(float f) {
    unsigned int u = __float_as_uint(f);
    unsigned int r = u + 0x7FFFu + ((u >> 16) & 1u);   // round-to-nearest-even
    return (unsigned short)(r >> 16);
}
__device__ __forceinline__ float bf2f(unsigned short h) {
    return __uint_as_float(((unsigned int)h) << 16);
}

// Load A fragment (16x32 bf16, ISA 7.12.2 layout) from a row-major tile.
// arow must already be tile_base + m*stride + half*8, caller adds kb.
__device__ __forceinline__ v16bf load_afrag(const unsigned short* arow_kb) {
    v16bf a;
    reinterpret_cast<uint4*>(&a)[0] = *reinterpret_cast<const uint4*>(arow_kb);
    reinterpret_cast<uint4*>(&a)[1] = *reinterpret_cast<const uint4*>(arow_kb + 16);
    return a;
}
// Load B fragment (32x16 bf16) from transposed weights [Nout][K]:
// per lane two 8-K chunks at +0 and +16 (same layout as A fragment).
__device__ __forceinline__ v16bf load_bfrag(const unsigned short* brow_kb) {
    v16bf b;
    reinterpret_cast<uint4*>(&b)[0] = *reinterpret_cast<const uint4*>(brow_kb);
    reinterpret_cast<uint4*>(&b)[1] = *reinterpret_cast<const uint4*>(brow_kb + 16);
    return b;
}

#define WMMA_BF16(a, b, c) \
    __builtin_amdgcn_wmma_f32_16x16x32_bf16(false, (a), false, (b), (short)0, (c), false, false)

// ---------------- kernel 1: weight transpose + bf16 convert ----------------
__global__ void k_prep_weights(const float* __restrict__ We, const float* __restrict__ W0,
                               const float* __restrict__ W1,
                               unsigned short* __restrict__ WeT, unsigned short* __restrict__ W0T,
                               unsigned short* __restrict__ W1T) {
    int idx = blockIdx.x * 256 + threadIdx.x;        // 524288 total
    if (idx < 131072) {                              // WeT[256][512]
        int n = idx >> 9, k = idx & 511;
        WeT[idx] = f2bf(We[k * EDIM + n]);
    } else if (idx < 393216) {                       // W0T[512][512]
        int e = idx - 131072;
        int n = e >> 9, k = e & 511;
        W0T[e] = f2bf(W0[k * H0D + n]);
    } else {                                         // W1T[256][512]
        int e = idx - 393216;
        int n = e >> 9, k = e & 511;
        W1T[e] = f2bf(W1[k * H1D + n]);
    }
}

// ---------------- kernel 2: feature -> bf16 (padded) ----------------
__global__ void k_prep_feat(const float* __restrict__ feature, unsigned short* __restrict__ fbf) {
    int idx = blockIdx.x * 256 + threadIdx.x;        // NP*512
    int row = idx >> 9;
    fbf[idx] = (row < NN) ? f2bf(feature[idx]) : (unsigned short)0;
}

// ---------------- kernel 3/9: WMMA GEMM, K=512, Nout=256, 1 mtile/block ------
// tgt==nullptr: emb mode -> out[row] = relu(...), emb_bf copy.
// tgt!=nullptr: scatter mode -> out[tgt[row]] = relu(...) when tgt>=0.
__global__ void k_gemm_embed(const unsigned short* __restrict__ A,   // [NP][512] bf16
                             const unsigned short* __restrict__ BT,  // [256][512] bf16
                             const float* __restrict__ bias,         // [256]
                             float* __restrict__ out,                // [NN][256]
                             unsigned short* __restrict__ emb_bf,    // [NP][256] or null
                             const int* __restrict__ tgt) {          // [NN] or null
    __shared__ unsigned short aTile[16 * 512];       // 16 KB
    const int mtile = blockIdx.x;
    const int tid = threadIdx.x;
    // stage 16x512 A tile (contiguous in memory)
    {
        const uint4* src = reinterpret_cast<const uint4*>(A + mtile * 16 * 512);
        uint4* dst = reinterpret_cast<uint4*>(aTile);
#pragma unroll
        for (int j = 0; j < 4; ++j) dst[tid + j * 256] = src[tid + j * 256];
    }
    __syncthreads();

    const int wave = tid >> 5, lane = tid & 31;
    const int n = lane & 15, half = lane >> 4;
    const unsigned short* arow = aTile + n * 512 + half * 8;     // m == lane&15
    const unsigned short* br0 = BT + ((wave * 2 + 0) * 16 + n) * 512 + half * 8;
    const unsigned short* br1 = BT + ((wave * 2 + 1) * 16 + n) * 512 + half * 8;
    v8f acc0 = {0.f, 0.f, 0.f, 0.f, 0.f, 0.f, 0.f, 0.f};
    v8f acc1 = acc0;
    for (int kb = 0; kb < 512; kb += 32) {
        v16bf a = load_afrag(arow + kb);
        acc0 = WMMA_BF16(a, load_bfrag(br0 + kb), acc0);
        acc1 = WMMA_BF16(a, load_bfrag(br1 + kb), acc1);
    }
    v8f accs[2] = {acc0, acc1};
#pragma unroll
    for (int ti = 0; ti < 2; ++ti) {
        const int col = (wave * 2 + ti) * 16 + n;
        const float bv = bias[col];
#pragma unroll
        for (int r = 0; r < 8; ++r) {
            int row = mtile * 16 + r + half * 8;
            float v = accs[ti][r] + bv;
            v = v > 0.f ? v : 0.f;
            if (row < NN) {
                if (tgt) {
                    int tg = tgt[row];
                    if (tg >= 0) out[tg * EDIM + col] = v;
                } else {
                    out[row * EDIM + col] = v;
                    emb_bf[row * EDIM + col] = f2bf(v);
                }
            }
        }
    }
}

// ---------------- kernel 4: fused edge MLP, 32 rows/block --------------------
__global__ void k_edge_mlp(const unsigned short* __restrict__ emb_bf,
                           const int* __restrict__ esrc, const int* __restrict__ edst,
                           const unsigned short* __restrict__ W0T, const float* __restrict__ b0,
                           const unsigned short* __restrict__ W1T, const float* __restrict__ b1,
                           const float* __restrict__ Wlk, const float* __restrict__ blk,
                           unsigned short* __restrict__ h1g, float* __restrict__ lk) {
    __shared__ unsigned short sbuf[32 * 512];        // 32 KB: s rows
    __shared__ unsigned short h0buf[32 * 512];       // 32 KB: h0 rows
    __shared__ float lkl[32];
    const int tid = threadIdx.x;
    const int tbase = blockIdx.x * 32;

    // ---- build s tile: [emb[seg] | emb[dst] or 0], 8 threads per row ----
    {
        int r = tid >> 3;                            // 0..31
        int c0 = (tid & 7) * 64;                     // 64 bf16 per thread
        int t = tbase + r;
        int sidx = -1, didx = -1;
        if (t < NE) { sidx = esrc[t]; didx = edst[t]; }
        else if (t < TT) { sidx = t - NE; }
        uint4* drow = reinterpret_cast<uint4*>(sbuf + r * 512 + c0);
#pragma unroll
        for (int e = 0; e < 8; ++e) {
            int col = c0 + e * 8;
            uint4 v = make_uint4(0u, 0u, 0u, 0u);
            if (col < 256) {
                if (sidx >= 0) v = *reinterpret_cast<const uint4*>(emb_bf + sidx * EDIM + col);
            } else {
                if (didx >= 0) v = *reinterpret_cast<const uint4*>(emb_bf + didx * EDIM + (col - 256));
            }
            drow[e] = v;
        }
        if (tid < 32) lkl[tid] = 0.f;
    }
    __syncthreads();

    const int wave = tid >> 5, lane = tid & 31;
    const int n = lane & 15, half = lane >> 4;

    // ---- layer 0: h0 = relu(s @ W0 + b0); 2 mtiles x 4 ntiles per wave ----
    {
        const unsigned short* a0row = sbuf + n * 512 + half * 8;          // rows 0..15
        const unsigned short* a1row = sbuf + (16 + n) * 512 + half * 8;   // rows 16..31
        v8f acc[8];
#pragma unroll
        for (int i = 0; i < 8; ++i) acc[i] = (v8f){0.f,0.f,0.f,0.f,0.f,0.f,0.f,0.f};
        for (int kb = 0; kb < 512; kb += 32) {
            v16bf a0 = load_afrag(a0row + kb);
            v16bf a1 = load_afrag(a1row + kb);
#pragma unroll
            for (int ti = 0; ti < 4; ++ti) {
                const unsigned short* br =
                    W0T + ((wave * 4 + ti) * 16 + n) * 512 + half * 8 + kb;
                v16bf b = load_bfrag(br);
                acc[ti * 2 + 0] = WMMA_BF16(a0, b, acc[ti * 2 + 0]);
                acc[ti * 2 + 1] = WMMA_BF16(a1, b, acc[ti * 2 + 1]);
            }
        }
#pragma unroll
        for (int ti = 0; ti < 4; ++ti) {
            int col = (wave * 4 + ti) * 16 + n;
            float bv = b0[col];
#pragma unroll
            for (int m = 0; m < 2; ++m) {
#pragma unroll
                for (int r = 0; r < 8; ++r) {
                    float v = acc[ti * 2 + m][r] + bv;
                    v = v > 0.f ? v : 0.f;
                    h0buf[(m * 16 + r + half * 8) * 512 + col] = f2bf(v);
                }
            }
        }
    }
    __syncthreads();

    // ---- layer 1: h1 = relu(h0 @ W1 + b1); 2 mtiles x 2 ntiles per wave -----
    {
        const unsigned short* a0row = h0buf + n * 512 + half * 8;
        const unsigned short* a1row = h0buf + (16 + n) * 512 + half * 8;
        v8f acc[4];
#pragma unroll
        for (int i = 0; i < 4; ++i) acc[i] = (v8f){0.f,0.f,0.f,0.f,0.f,0.f,0.f,0.f};
        for (int kb = 0; kb < 512; kb += 32) {
            v16bf a0 = load_afrag(a0row + kb);
            v16bf a1 = load_afrag(a1row + kb);
#pragma unroll
            for (int ti = 0; ti < 2; ++ti) {
                const unsigned short* br =
                    W1T + ((wave * 2 + ti) * 16 + n) * 512 + half * 8 + kb;
                v16bf b = load_bfrag(br);
                acc[ti * 2 + 0] = WMMA_BF16(a0, b, acc[ti * 2 + 0]);
                acc[ti * 2 + 1] = WMMA_BF16(a1, b, acc[ti * 2 + 1]);
            }
        }
#pragma unroll
        for (int ti = 0; ti < 2; ++ti) {
            int col = (wave * 2 + ti) * 16 + n;
            float bv = b1[col];
            float wl = Wlk[col];
#pragma unroll
            for (int m = 0; m < 2; ++m) {
#pragma unroll
                for (int r = 0; r < 8; ++r) {
                    float v = acc[ti * 2 + m][r] + bv;
                    v = v > 0.f ? v : 0.f;
                    int row = m * 16 + r + half * 8;
                    int t = tbase + row;
                    if (t < TT) h1g[(size_t)t * H1D + col] = f2bf(v);
                    atomicAdd(&lkl[row], v * wl);
                }
            }
        }
    }
    __syncthreads();
    if (tid < 32) {
        int t = tbase + tid;
        if (t < TT) lk[t] = lkl[tid] + blk[0];
    }
}

// ---------------- kernel 5/6: packed segment-argmax ----------------
__global__ void k_init_packed(unsigned long long* __restrict__ packed) {
    int i = blockIdx.x * 256 + threadIdx.x;
    if (i < NN) packed[i] = 0ull;
}
__global__ void k_segmax(const float* __restrict__ lk, const int* __restrict__ esrc,
                         unsigned long long* __restrict__ packed) {
    int t = blockIdx.x * 256 + threadIdx.x;
    if (t >= TT) return;
    int i = (t < NE) ? esrc[t] : (t - NE);
    unsigned int b = __float_as_uint(lk[t]);
    unsigned int mkey = (b & 0x80000000u) ? ~b : (b | 0x80000000u);  // order-preserving
    unsigned long long key = ((unsigned long long)mkey << 32) | (unsigned int)t;
    atomicMax(&packed[i], key);   // max lk, ties -> max t (matches reference)
}

// ---------------- kernel 7: action head + target decision ----------------
__global__ void k_decide(const unsigned long long* __restrict__ packed,
                         const int* __restrict__ edst, const unsigned short* __restrict__ h1g,
                         const float* __restrict__ Wa, const float* __restrict__ ba,
                         int* __restrict__ tgt) {
    int i = blockIdx.x * 256 + threadIdx.x;
    if (i >= NN) return;
    int twin = (int)(packed[i] & 0xFFFFFFFFull);
    int chosen = (twin < NE) ? edst[twin] : -1;
    float l0 = ba[0], l1 = ba[1];
    const unsigned short* h = h1g + (size_t)twin * H1D;
    for (int j = 0; j < H1D; ++j) {
        float hv = bf2f(h[j]);
        l0 += hv * Wa[2 * j];
        l1 += hv * Wa[2 * j + 1];
    }
    // argmax: at==1 only when l1 strictly greater (argmax picks first on tie)
    tgt[i] = (chosen >= 0 && l1 > l0) ? chosen : -1;
}

// ---------------- kernel 8: mean feature (bf16, padded) ----------------
__global__ void k_meanfeat(const float* __restrict__ feature, const int* __restrict__ tgt,
                           unsigned short* __restrict__ mf) {
    int idx = blockIdx.x * 256 + threadIdx.x;        // NP*512
    int row = idx >> 9, k = idx & 511;
    unsigned short v = 0;
    if (row < NN) {
        int tg = tgt[row];
        if (tg >= 0) v = f2bf(0.5f * (feature[row * FIN + k] + feature[tg * FIN + k]));
    }
    mf[idx] = v;
}

// ---------------- launch ----------------
extern "C" void kernel_launch(void* const* d_in, const int* in_sizes, int n_in,
                              void* d_out, int out_size, void* d_ws, size_t ws_size,
                              hipStream_t stream) {
    (void)in_sizes; (void)n_in; (void)out_size; (void)ws_size;
    const float* feature = (const float*)d_in[0];
    const float* We      = (const float*)d_in[1];
    const float* be      = (const float*)d_in[2];
    const float* W0      = (const float*)d_in[3];
    const float* b0      = (const float*)d_in[4];
    const float* W1      = (const float*)d_in[5];
    const float* b1      = (const float*)d_in[6];
    const float* Wlk     = (const float*)d_in[7];
    const float* blk     = (const float*)d_in[8];
    const float* Wa      = (const float*)d_in[9];
    const float* ba      = (const float*)d_in[10];
    const int* esrc      = (const int*)d_in[11];
    const int* edst      = (const int*)d_in[12];
    float* out           = (float*)d_out;

    // workspace carve-up (256B aligned)
    unsigned char* w = (unsigned char*)d_ws;
    size_t off = 0;
    auto take = [&](size_t bytes) -> unsigned char* {
        unsigned char* p = w + off;
        off = (off + bytes + 255) & ~(size_t)255;
        return p;
    };
    unsigned short* feat_bf = (unsigned short*)take((size_t)NP * FIN * 2);
    unsigned short* mf_bf   = (unsigned short*)take((size_t)NP * FIN * 2);
    unsigned short* WeT     = (unsigned short*)take((size_t)EDIM * FIN * 2);
    unsigned short* W0T     = (unsigned short*)take((size_t)H0D * 512 * 2);
    unsigned short* W1T     = (unsigned short*)take((size_t)H1D * H0D * 2);
    unsigned short* emb_bf  = (unsigned short*)take((size_t)NP * EDIM * 2);
    unsigned short* h1g     = (unsigned short*)take((size_t)TP * H1D * 2);
    float*          lk      = (float*)take((size_t)TP * 4);
    unsigned long long* packed = (unsigned long long*)take((size_t)NN * 8);
    int*            tgt     = (int*)take((size_t)NN * 4);

    k_prep_weights<<<2048, 256, 0, stream>>>(We, W0, W1, WeT, W0T, W1T);
    k_prep_feat<<<(NP * FIN) / 256, 256, 0, stream>>>(feature, feat_bf);
    k_gemm_embed<<<NP / 16, 256, 0, stream>>>(feat_bf, WeT, be, out, emb_bf, nullptr);
    k_edge_mlp<<<TP / 32, 256, 0, stream>>>(emb_bf, esrc, edst, W0T, b0, W1T, b1, Wlk, blk,
                                            h1g, lk);
    k_init_packed<<<(NN + 255) / 256, 256, 0, stream>>>(packed);
    k_segmax<<<(TT + 255) / 256, 256, 0, stream>>>(lk, esrc, packed);
    k_decide<<<(NN + 255) / 256, 256, 0, stream>>>(packed, edst, h1g, Wa, ba, tgt);
    k_meanfeat<<<(NP * FIN) / 256, 256, 0, stream>>>(feature, tgt, mf_bf);
    k_gemm_embed<<<NP / 16, 256, 0, stream>>>(mf_bf, WeT, be, out, nullptr, tgt);
}
